// TimeMixerBlock_49194555408698
// MI455X (gfx1250) — matrix operand order repaired and checked by reading, not compiled, not run on hardware
//
#include <hip/hip_runtime.h>
#include <hip/hip_bf16.h>

typedef __bf16 bf16_t;
typedef __attribute__((ext_vector_type(16))) __bf16 v16bf;
typedef __attribute__((ext_vector_type(8)))  __bf16 v8bf;
typedef __attribute__((ext_vector_type(4)))  __bf16 v4bf;
typedef __attribute__((ext_vector_type(8)))  float  v8f;
typedef __attribute__((ext_vector_type(4)))  float  v4f;

#define T_DIM   512
#define ROWS_WG 64
#define THREADS 256

// ---------------------------------------------------------------------------
// Prep: W (f32, row-major [s][t]) -> lower-triangular-masked bf16.
// ---------------------------------------------------------------------------
__global__ __launch_bounds__(THREADS)
void prep_w_kernel(const float* __restrict__ W, bf16_t* __restrict__ Wbf) {
  int idx  = blockIdx.x * blockDim.x + threadIdx.x;   // 0 .. 512*512-1
  int srow = idx >> 9;
  int t    = idx & (T_DIM - 1);
  float v  = (t <= srow) ? W[idx] : 0.0f;
  Wbf[idx] = (bf16_t)v;
}

// ---------------------------------------------------------------------------
// Triangular GEMM: acc[i][j] += tril(W)[sblk(i)] x tile[rows j].
// Wave w owns s-blocks  s = i*128 + w*16  (i = 0..3) -> balanced across waves.
// `wave` is readfirstlane'd into an SGPR, so all diagonal-skip guards compile
// to scalar compares + s_cbranch (EXEC never modified; WMMA sees all-ones).
// A fragment (16x32 bf16): lane m=L%16, half=L>>4; elems[0..15]=K(k0+half*16..+15)
// B fragment (32x16 bf16): lane n=L%16, half=L>>4; elems[0..15]=K(k0+half*16..+15)
// ---------------------------------------------------------------------------
__device__ __forceinline__
void tri_gemm(const bf16_t* __restrict__ Wbf, const bf16_t* tile,
              int wave, int lane, v8f acc[4][4]) {
  const int m    = lane & 15;
  const int hl   = lane >> 4;
  const int kmax = 3 * 128 + wave * 16 + 16;          // scalar: last active K-block
  for (int k0 = 0; k0 < kmax; k0 += 32) {
    v16bf bfr[4];
#pragma unroll
    for (int j = 0; j < 4; ++j) {
      const bf16_t* p = tile + (j * 16 + m) * T_DIM + k0 + hl * 16;
      bfr[j] = *(const v16bf*)p;                      // 32B contiguous LDS read
    }
#pragma unroll
    for (int i = 0; i < 4; ++i) {
      if (k0 < i * 128 + wave * 16 + 16) {            // scalar condition (SGPR wave)
        int srow = i * 128 + wave * 16 + m;
        const bf16_t* p = Wbf + (size_t)srow * T_DIM + k0 + hl * 16;
        v8bf lo = *(const v8bf*)p;
        v8bf hi = *(const v8bf*)(p + 8);
        v16bf a = __builtin_shufflevector(lo, hi, 0,1,2,3,4,5,6,7,
                                                  8,9,10,11,12,13,14,15);
#pragma unroll
        for (int j = 0; j < 4; ++j)
          acc[i][j] = __builtin_amdgcn_wmma_f32_16x16x32_bf16(
              false, a, false, bfr[j], (short)0, acc[i][j], false, false);
      }
    }
  }
}

// ---------------------------------------------------------------------------
// Fused: out = tril(W2) * gelu(tril(W1) * x + b1) + b2, per 64-row tile.
// Single 64KB LDS buffer reused: x(bf16) -> GEMM1 -> gelu(h)(bf16) -> GEMM2.
// ---------------------------------------------------------------------------
__global__ __launch_bounds__(THREADS)
void triu_mlp_kernel(const float* __restrict__ x,
                     const bf16_t* __restrict__ W1bf,
                     const float* __restrict__ b1,
                     const bf16_t* __restrict__ W2bf,
                     const float* __restrict__ b2,
                     float* __restrict__ out) {
  __shared__ bf16_t tile[ROWS_WG * T_DIM];            // 64 KB

  const int tid  = threadIdx.x;
  const int lane = tid & 31;
  // Force wave index into an SGPR: it is uniform within a wave, and making it
  // scalar turns all triangular-skip guards into SALU branches (EXEC untouched).
  const int wave = __builtin_amdgcn_readfirstlane(tid >> 5);   // 0..7
  const int m    = lane & 15;
  const int hl   = lane >> 4;
  const size_t rowBase = (size_t)blockIdx.x * ROWS_WG;

  // ---- stage x tile (f32 global -> bf16 LDS), coalesced float4 loads ----
  {
    const float* src = x + rowBase * T_DIM;
    for (int i = tid * 4; i < ROWS_WG * T_DIM; i += THREADS * 4) {
      v4f v = *(const v4f*)(src + i);
      v4bf b; b.x = (bf16_t)v.x; b.y = (bf16_t)v.y;
              b.z = (bf16_t)v.z; b.w = (bf16_t)v.w;
      *(v4bf*)&tile[i] = b;                           // 8B LDS store
    }
  }
  __syncthreads();

  v8f acc[4][4];
#pragma unroll
  for (int i = 0; i < 4; ++i)
#pragma unroll
    for (int j = 0; j < 4; ++j) acc[i][j] = (v8f){};

  // ---- GEMM1: h^T[s, row] tiles ----
  tri_gemm(W1bf, tile, wave, lane, acc);
  __syncthreads();                                    // all waves done reading x

  // ---- epilogue 1: bias + exact-erf GELU, write bf16 g back into LDS ----
#pragma unroll
  for (int i = 0; i < 4; ++i) {
    int sbase = i * 128 + wave * 16 + hl * 8;         // D: M = r + 8*hl
#pragma unroll
    for (int j = 0; j < 4; ++j) {
      int row = j * 16 + m;                           // D: N = lane%16
      v8bf g;
#pragma unroll
      for (int r = 0; r < 8; ++r) {
        float h = acc[i][j][r] + b1[sbase + r];
        float gv = 0.5f * h * (1.0f + erff(h * 0.70710678118654752f));
        g[r] = (bf16_t)gv;
      }
      *(v8bf*)&tile[row * T_DIM + sbase] = g;         // 16B LDS store
      acc[i][j] = (v8f){};
    }
  }
  __syncthreads();

  // ---- GEMM2: out^T[u, row] tiles ----
  tri_gemm(W2bf, tile, wave, lane, acc);

  // ---- epilogue 2: bias + f32 store (8 consecutive floats per lane) ----
#pragma unroll
  for (int i = 0; i < 4; ++i) {
    int ubase = i * 128 + wave * 16 + hl * 8;
#pragma unroll
    for (int j = 0; j < 4; ++j) {
      int row = j * 16 + m;
      float* dst = out + (rowBase + row) * T_DIM + ubase;
      v4f lo, hi;
#pragma unroll
      for (int r = 0; r < 4; ++r) {
        lo[r] = acc[i][j][r]     + b2[ubase + r];
        hi[r] = acc[i][j][r + 4] + b2[ubase + r + 4];
      }
      *(v4f*)dst       = lo;
      *(v4f*)(dst + 4) = hi;
    }
  }
}

// ---------------------------------------------------------------------------
extern "C" void kernel_launch(void* const* d_in, const int* in_sizes, int n_in,
                              void* d_out, int out_size, void* d_ws, size_t ws_size,
                              hipStream_t stream) {
  const float* x  = (const float*)d_in[0];
  const float* W1 = (const float*)d_in[1];
  const float* b1 = (const float*)d_in[2];
  const float* W2 = (const float*)d_in[3];
  const float* b2 = (const float*)d_in[4];
  float* out = (float*)d_out;

  bf16_t* W1bf = (bf16_t*)d_ws;                       // 512*512*2 B
  bf16_t* W2bf = W1bf + T_DIM * T_DIM;                // next 512 KB

  const int wElems = T_DIM * T_DIM;
  prep_w_kernel<<<wElems / THREADS, THREADS, 0, stream>>>(W1, W1bf);
  prep_w_kernel<<<wElems / THREADS, THREADS, 0, stream>>>(W2, W2bf);

  const int nRows  = 64 * 1024;                       // B*C
  triu_mlp_kernel<<<nRows / ROWS_WG, THREADS, 0, stream>>>(
      x, W1bf, b1, W2bf, b2, out);
}
